// Net_32169305047431
// MI455X (gfx1250) — matrix operand order repaired and checked by reading, not compiled, not run on hardware
//
#include <hip/hip_runtime.h>

typedef __attribute__((ext_vector_type(16))) _Float16 v16h;
typedef __attribute__((ext_vector_type(8)))  float    v8f;

#define NPTS  65536
#define KNN   16
#define CH    64
#define NEDGE (NPTS*KNN)
#define EPSB  1e-5f

// ---------------- WMMA fragment helpers (ISA 7.12.2 layouts, wave32) -------

// A fragment: rows [row0..row0+15], K = [k0..k0+31] of a row-major [*,64] f32
// matrix, optionally fused BN(scale,shift)+ReLU, converted to f16.
// lane in [0,32): m = lane&15, half = lane>>4.
// VGPR v: K = k0 + (v>>2)*16 + half*8 + (v&3)*2  (two packed halves).
template <bool BN>
__device__ __forceinline__ v16h load_a_frag(const float* __restrict__ X, int row0,
                                            int k0, int lane,
                                            const float* __restrict__ scl,
                                            const float* __restrict__ shf) {
  int half = lane >> 4, m = lane & 15;
  const float* rp = X + (size_t)(row0 + m) * CH;
  v16h a;
#pragma unroll
  for (int v = 0; v < 8; ++v) {
    int kk = k0 + ((v >> 2) << 4) + (half << 3) + ((v & 3) << 1);
    float x0 = rp[kk];
    float x1 = rp[kk + 1];
    if (BN) {
      x0 = fmaxf(0.f, fmaf(x0, scl[kk], shf[kk]));
      x1 = fmaxf(0.f, fmaf(x1, scl[kk + 1], shf[kk + 1]));
    }
    a[2 * v]     = (_Float16)x0;
    a[2 * v + 1] = (_Float16)x1;
  }
  return a;
}

// B fragment: B[k][n] = W[n][k] for W row-major [64][64] (co,ci).
// lane: n = n0 + (lane&15); VGPR v: K = (v>>2)*16 + half*8 + (v&3)*2 (+0,+1).
__device__ __forceinline__ v16h load_b_frag(const float* __restrict__ W, int n0,
                                            int k0, int lane) {
  int n = n0 + (lane & 15), hs = lane >> 4;
  const float* rp = W + (size_t)n * CH + k0 + (hs << 3);
  v16h b;
#pragma unroll
  for (int v = 0; v < 8; ++v) {
    b[2 * v]     = (_Float16)rp[2 * v + ((v >> 2) << 3)];
    b[2 * v + 1] = (_Float16)rp[2 * v + 1 + ((v >> 2) << 3)];
  }
  return b;
}

// ---------------- K1/K12: T = act(X) @ W^T  (+ channel stats) --------------
template <bool BN>
__global__ __launch_bounds__(256) void gemm64(const float* __restrict__ X,
                                              const float* __restrict__ W,
                                              float* __restrict__ T,
                                              const float* __restrict__ aScl,
                                              const float* __restrict__ aShf,
                                              double* __restrict__ gSum,
                                              double* __restrict__ gSq) {
  __shared__ float sSum[CH], sSq[CH];
  int tid = threadIdx.x, lane = tid & 31, wave = tid >> 5, half = lane >> 4;
  if (tid < CH) { sSum[tid] = 0.f; sSq[tid] = 0.f; }
  __syncthreads();
  int row0 = blockIdx.x * 128 + wave * 16;
  v16h a0 = load_a_frag<BN>(X, row0, 0, lane, aScl, aShf);
  v16h a1 = load_a_frag<BN>(X, row0, 32, lane, aScl, aShf);
#pragma unroll
  for (int nt = 0; nt < 4; ++nt) {
    v16h b0 = load_b_frag(W, nt * 16, 0, lane);
    v16h b1 = load_b_frag(W, nt * 16, 32, lane);
    v8f c = {};
    c = __builtin_amdgcn_wmma_f32_16x16x32_f16(false, a0, false, b0, (short)0, c, false, false);
    c = __builtin_amdgcn_wmma_f32_16x16x32_f16(false, a1, false, b1, (short)0, c, false, false);
    int ncol = nt * 16 + (lane & 15);
    float ps = 0.f, pq = 0.f;
#pragma unroll
    for (int r = 0; r < 8; ++r) {
      float val = c[r];
      T[(size_t)(row0 + r + 8 * half) * CH + ncol] = val;
      ps += val;
      pq = fmaf(val, val, pq);
    }
    // lanes l and l+16 own the same output column: fold halves, one atomic.
    ps += __shfl_xor(ps, 16, 32);
    pq += __shfl_xor(pq, 16, 32);
    if (half == 0) { atomicAdd(&sSum[ncol], ps); atomicAdd(&sSq[ncol], pq); }
  }
  __syncthreads();
  if (tid < CH) {
    atomicAdd(&gSum[tid], (double)sSum[tid]);
    atomicAdd(&gSq[tid],  (double)sSq[tid]);
  }
}

// ---------------- K3: h = relu(bn1(t)); a_src/a_dst/hW = h@{src,dst,lin}W^T+b
__global__ __launch_bounds__(256) void gemm64_triple(
    const float* __restrict__ T,
    const float* __restrict__ bn1s, const float* __restrict__ bn1h,
    const float* __restrict__ Wsrc, const float* __restrict__ Bsrc,
    const float* __restrict__ Wdst, const float* __restrict__ Bdst,
    const float* __restrict__ Wlin, const float* __restrict__ Blin,
    float* __restrict__ Asrc, float* __restrict__ Adst, float* __restrict__ HW) {
  int tid = threadIdx.x, lane = tid & 31, wave = tid >> 5, half = lane >> 4;
  int row0 = blockIdx.x * 128 + wave * 16;
  v16h a0 = load_a_frag<true>(T, row0, 0, lane, bn1s, bn1h);
  v16h a1 = load_a_frag<true>(T, row0, 32, lane, bn1s, bn1h);
  const float* Ws[3] = {Wsrc, Wdst, Wlin};
  const float* Bs[3] = {Bsrc, Bdst, Blin};
  float* Os[3] = {Asrc, Adst, HW};
#pragma unroll
  for (int m = 0; m < 3; ++m) {
#pragma unroll
    for (int nt = 0; nt < 4; ++nt) {
      v16h b0 = load_b_frag(Ws[m], nt * 16, 0, lane);
      v16h b1 = load_b_frag(Ws[m], nt * 16, 32, lane);
      v8f c = {};
      c = __builtin_amdgcn_wmma_f32_16x16x32_f16(false, a0, false, b0, (short)0, c, false, false);
      c = __builtin_amdgcn_wmma_f32_16x16x32_f16(false, a1, false, b1, (short)0, c, false, false);
      int ncol = nt * 16 + (lane & 15);
      float bias = Bs[m][ncol];
#pragma unroll
      for (int r = 0; r < 8; ++r)
        Os[m][(size_t)(row0 + r + 8 * half) * CH + ncol] = c[r] + bias;
    }
  }
}

// ---------------- K4: u = (pos[src]-pos[dst])@pw1^T + pb1 ; stats over E ----
__global__ __launch_bounds__(256) void pos_stats(const int* __restrict__ src,
                                                 const float* __restrict__ pos,
                                                 const float* __restrict__ pw1,
                                                 const float* __restrict__ pb1,
                                                 float* __restrict__ ubuf,
                                                 double* __restrict__ gSum,
                                                 double* __restrict__ gSq) {
  __shared__ float s[6];
  if (threadIdx.x < 6) s[threadIdx.x] = 0.f;
  __syncthreads();
  int e = blockIdx.x * 256 + threadIdx.x;
  int si = src[e], di = e >> 4;
  float r0 = pos[si * 3 + 0] - pos[di * 3 + 0];
  float r1 = pos[si * 3 + 1] - pos[di * 3 + 1];
  float r2 = pos[si * 3 + 2] - pos[di * 3 + 2];
  float u[3];
#pragma unroll
  for (int j = 0; j < 3; ++j) {
    u[j] = fmaf(r0, pw1[j * 3 + 0], fmaf(r1, pw1[j * 3 + 1], fmaf(r2, pw1[j * 3 + 2], pb1[j])));
    ubuf[(size_t)e * 3 + j] = u[j];
  }
#pragma unroll
  for (int j = 0; j < 3; ++j) {
    float v = u[j], q = u[j] * u[j];
#pragma unroll
    for (int o = 16; o > 0; o >>= 1) { v += __shfl_xor(v, o, 32); q += __shfl_xor(q, o, 32); }
    if ((threadIdx.x & 31) == 0) { atomicAdd(&s[j], v); atomicAdd(&s[3 + j], q); }
  }
  __syncthreads();
  if (threadIdx.x < 3) {
    atomicAdd(&gSum[threadIdx.x], (double)s[threadIdx.x]);
    atomicAdd(&gSq[threadIdx.x],  (double)s[3 + threadIdx.x]);
  }
}

// ---------------- BN finalize: scale/shift from sums ------------------------
__global__ void bn_finalize(const double* __restrict__ sum, const double* __restrict__ sq,
                            const float* __restrict__ g, const float* __restrict__ b,
                            float* __restrict__ scl, float* __restrict__ shf,
                            int nch, double invN) {
  int i = threadIdx.x;
  if (i < nch) {
    double mu  = sum[i] * invN;
    double var = sq[i] * invN - mu * mu;
    float  s   = g[i] * rsqrtf((float)var + EPSB);
    scl[i] = s;
    shf[i] = fmaf(-(float)mu, s, b[i]);
  }
}

// ---------------- K6: stats of a = a_src[src]-a_dst[dst]+delta over E -------
__global__ __launch_bounds__(256) void attn1_stats(
    const int* __restrict__ src,
    const float* __restrict__ Asrc, const float* __restrict__ Adst,
    const float* __restrict__ ubuf,
    const float* __restrict__ posS, const float* __restrict__ posH,
    const float* __restrict__ pw2, const float* __restrict__ pb2,
    double* __restrict__ gSum, double* __restrict__ gSq) {
  __shared__ float sAcc[512];
  int tid = threadIdx.x, c = tid & 63, j = tid >> 6;
  float w0 = pw2[c * 3], w1 = pw2[c * 3 + 1], w2 = pw2[c * 3 + 2], bc = pb2[c];
  float ps0 = posS[0], ps1 = posS[1], ps2 = posS[2];
  float ph0 = posH[0], ph1 = posH[1], ph2 = posH[2];
  float acc = 0.f, accq = 0.f;
  int base = blockIdx.x * 256;
  for (int it = j; it < 256; it += 4) {
    int e = base + it;
    int si = src[e];
    float u0 = ubuf[(size_t)e * 3], u1 = ubuf[(size_t)e * 3 + 1], u2 = ubuf[(size_t)e * 3 + 2];
    float d0 = fmaxf(0.f, fmaf(u0, ps0, ph0));
    float d1 = fmaxf(0.f, fmaf(u1, ps1, ph1));
    float d2 = fmaxf(0.f, fmaf(u2, ps2, ph2));
    float delta = fmaf(d0, w0, fmaf(d1, w1, fmaf(d2, w2, bc)));
    float a = Asrc[(size_t)si * CH + c] - Adst[(size_t)(e >> 4) * CH + c] + delta;
    acc += a; accq = fmaf(a, a, accq);
  }
  sAcc[tid] = acc; sAcc[256 + tid] = accq;
  __syncthreads();
  if (tid < 64) {
    float s = sAcc[tid] + sAcc[tid + 64] + sAcc[tid + 128] + sAcc[tid + 192];
    float q = sAcc[256 + tid] + sAcc[320 + tid] + sAcc[384 + tid] + sAcc[448 + tid];
    atomicAdd(&gSum[tid], (double)s);
    atomicAdd(&gSq[tid],  (double)q);
  }
}

// ---------------- K8: y = relu(bn(a)) @ w1^T + b1 ; stats over E ------------
__global__ __launch_bounds__(256) void attn_mlp1(
    const int* __restrict__ src,
    const float* __restrict__ Asrc, const float* __restrict__ Adst,
    const float* __restrict__ ubuf,
    const float* __restrict__ posS, const float* __restrict__ posH,
    const float* __restrict__ pw2, const float* __restrict__ pb2,
    const float* __restrict__ a1s, const float* __restrict__ a1h,
    const float* __restrict__ w1, const float* __restrict__ b1,
    float* __restrict__ ybuf, double* __restrict__ gSum, double* __restrict__ gSq) {
  __shared__ float s[16];
  if (threadIdx.x < 16) s[threadIdx.x] = 0.f;
  __syncthreads();
  int e = blockIdx.x * 256 + threadIdx.x;
  int si = src[e], di = e >> 4;
  float u0 = ubuf[(size_t)e * 3], u1 = ubuf[(size_t)e * 3 + 1], u2 = ubuf[(size_t)e * 3 + 2];
  float d0 = fmaxf(0.f, fmaf(u0, posS[0], posH[0]));
  float d1 = fmaxf(0.f, fmaf(u1, posS[1], posH[1]));
  float d2 = fmaxf(0.f, fmaf(u2, posS[2], posH[2]));
  const float* as = Asrc + (size_t)si * CH;
  const float* ad = Adst + (size_t)di * CH;
  float y[8];
#pragma unroll
  for (int o = 0; o < 8; ++o) y[o] = b1[o];
#pragma unroll 8
  for (int c = 0; c < CH; ++c) {
    float delta = fmaf(d0, pw2[c * 3], fmaf(d1, pw2[c * 3 + 1], fmaf(d2, pw2[c * 3 + 2], pb2[c])));
    float v = fmaxf(0.f, fmaf(as[c] - ad[c] + delta, a1s[c], a1h[c]));
#pragma unroll
    for (int o = 0; o < 8; ++o) y[o] = fmaf(v, w1[o * CH + c], y[o]);
  }
#pragma unroll
  for (int o = 0; o < 8; ++o) {
    ybuf[(size_t)e * 8 + o] = y[o];
    float v = y[o], q = y[o] * y[o];
#pragma unroll
    for (int off = 16; off > 0; off >>= 1) { v += __shfl_xor(v, off, 32); q += __shfl_xor(q, off, 32); }
    if ((threadIdx.x & 31) == 0) { atomicAdd(&s[o], v); atomicAdd(&s[8 + o], q); }
  }
  __syncthreads();
  if (threadIdx.x < 8) {
    atomicAdd(&gSum[threadIdx.x], (double)s[threadIdx.x]);
    atomicAdd(&gSq[threadIdx.x],  (double)s[8 + threadIdx.x]);
  }
}

// ---------------- K10: z = relu(bn(y))@w2^T+b2 ; softmax(16) ; aggregate ----
__global__ __launch_bounds__(256) void softmax_agg(
    const int* __restrict__ src,
    const float* __restrict__ ybuf, const float* __restrict__ ubuf,
    const float* __restrict__ posS, const float* __restrict__ posH,
    const float* __restrict__ pw2, const float* __restrict__ pb2,
    const float* __restrict__ a2s, const float* __restrict__ a2h,
    const float* __restrict__ w2, const float* __restrict__ b2,
    const float* __restrict__ HW,
    float* __restrict__ outb, double* __restrict__ gSum, double* __restrict__ gSq) {
  __shared__ float zl[256 * 8];
  __shared__ float dl[256 * 3];
  __shared__ int   srcl[256];
  __shared__ float sSum[CH], sSq[CH];
  int tid = threadIdx.x;
  if (tid < CH) { sSum[tid] = 0.f; sSq[tid] = 0.f; }
  int e0 = blockIdx.x * 256;
  {  // phase A: per-edge z, cache src & pos features
    int e = e0 + tid;
    int sidx = src[e];
    srcl[tid] = sidx;
    // warm the hW row for phase C (consumed after two barriers)
    __builtin_prefetch(&HW[(size_t)sidx * CH], 0, 0);
    float u0 = ubuf[(size_t)e * 3], u1 = ubuf[(size_t)e * 3 + 1], u2 = ubuf[(size_t)e * 3 + 2];
    dl[tid * 3 + 0] = fmaxf(0.f, fmaf(u0, posS[0], posH[0]));
    dl[tid * 3 + 1] = fmaxf(0.f, fmaf(u1, posS[1], posH[1]));
    dl[tid * 3 + 2] = fmaxf(0.f, fmaf(u2, posS[2], posH[2]));
    float z[8];
#pragma unroll
    for (int o = 0; o < 8; ++o) z[o] = b2[o];
#pragma unroll
    for (int k = 0; k < 8; ++k) {
      float v = fmaxf(0.f, fmaf(ybuf[(size_t)e * 8 + k], a2s[k], a2h[k]));
#pragma unroll
      for (int o = 0; o < 8; ++o) z[o] = fmaf(v, w2[o * 8 + k], z[o]);
    }
#pragma unroll
    for (int o = 0; o < 8; ++o) zl[tid * 8 + o] = z[o];
  }
  __syncthreads();
  if (tid < 128) {  // phase B: softmax over each node's 16 edges, per channel
    int node = tid >> 3, o = tid & 7;
    float m = -3.4e38f;
    for (int i = 0; i < 16; ++i) m = fmaxf(m, zl[(node * 16 + i) * 8 + o]);
    float ssum = 0.f; float ex[16];
    for (int i = 0; i < 16; ++i) { ex[i] = __expf(zl[(node * 16 + i) * 8 + o] - m); ssum += ex[i]; }
    float inv = 1.f / ssum;
    for (int i = 0; i < 16; ++i) zl[(node * 16 + i) * 8 + o] = ex[i] * inv;
  }
  __syncthreads();
  {  // phase C: out[node][c] = sum_e alpha[e][c&7] * (hW[src[e]][c] + delta)
    int c = tid & 63, nl = tid >> 6, o = c & 7;
    float w0 = pw2[c * 3], w1c = pw2[c * 3 + 1], w2c = pw2[c * 3 + 2], bc = pb2[c];
    for (int node = nl; node < 16; node += 4) {
      float acc = 0.f;
#pragma unroll
      for (int i = 0; i < 16; ++i) {
        int el = node * 16 + i;
        float delta = fmaf(dl[el * 3], w0, fmaf(dl[el * 3 + 1], w1c, fmaf(dl[el * 3 + 2], w2c, bc)));
        acc = fmaf(zl[el * 8 + o], HW[(size_t)srcl[el] * CH + c] + delta, acc);
      }
      outb[(size_t)((e0 >> 4) + node) * CH + c] = acc;
      atomicAdd(&sSum[c], acc); atomicAdd(&sSq[c], acc * acc);
    }
  }
  __syncthreads();
  if (tid < CH) {
    atomicAdd(&gSum[tid], (double)sSum[tid]);
    atomicAdd(&gSq[tid],  (double)sSq[tid]);
  }
}

// ---------------- K14: out = relu(bn3(t3) + x_skip) -------------------------
__global__ void final_out(const float* __restrict__ t3, const float* __restrict__ x,
                          const float* __restrict__ s3, const float* __restrict__ h3,
                          float* __restrict__ out) {
  int i = blockIdx.x * 256 + threadIdx.x;
  int c = i & 63;
  out[i] = fmaxf(0.f, fmaf(t3[i], s3[c], h3[c]) + x[i]);
}

__global__ void zero_stats(double* st) {
  int i = blockIdx.x * 256 + threadIdx.x;
  if (i < 1024) st[i] = 0.0;
}

// ---------------------------- launcher --------------------------------------
extern "C" void kernel_launch(void* const* d_in, const int* in_sizes, int n_in,
                              void* d_out, int out_size, void* d_ws, size_t ws_size,
                              hipStream_t stream) {
  (void)in_sizes; (void)n_in; (void)out_size; (void)ws_size;
  const float* x        = (const float*)d_in[0];
  const float* pos      = (const float*)d_in[1];
  const int*   src      = (const int*)d_in[2];   // edge_index row 0 (dst[e]=e>>4)
  const float* W_in     = (const float*)d_in[3];
  const float* W_out    = (const float*)d_in[4];
  const float* pos_w1   = (const float*)d_in[5];
  const float* pos_b1   = (const float*)d_in[6];
  const float* pos_bn_g = (const float*)d_in[7];
  const float* pos_bn_b = (const float*)d_in[8];
  const float* pos_w2   = (const float*)d_in[9];
  const float* pos_b2   = (const float*)d_in[10];
  const float* a1_g     = (const float*)d_in[11];
  const float* a1_b     = (const float*)d_in[12];
  const float* attn_w1  = (const float*)d_in[13];
  const float* attn_b1  = (const float*)d_in[14];
  const float* a2_g     = (const float*)d_in[15];
  const float* a2_b     = (const float*)d_in[16];
  const float* attn_w2  = (const float*)d_in[17];
  const float* attn_b2  = (const float*)d_in[18];
  const float* lin_w    = (const float*)d_in[19];
  const float* lin_b    = (const float*)d_in[20];
  const float* src_w    = (const float*)d_in[21];
  const float* src_b    = (const float*)d_in[22];
  const float* dst_w    = (const float*)d_in[23];
  const float* dst_b    = (const float*)d_in[24];
  const float* bn1_g    = (const float*)d_in[25];
  const float* bn1_b    = (const float*)d_in[26];
  const float* bn2_g    = (const float*)d_in[27];
  const float* bn2_b    = (const float*)d_in[28];
  const float* bn3_g    = (const float*)d_in[29];
  const float* bn3_b    = (const float*)d_in[30];

  char* w = (char*)d_ws;
  double* stats  = (double*)w;              // 1024 doubles (8 KB)
  float*  params = (float*)(w + 8192);      // 1024 floats  (4 KB)
  float*  tbuf   = (float*)(w + 16384);                 // N*64
  float*  asrcb  = tbuf  + (size_t)NPTS * CH;           // N*64
  float*  adstb  = asrcb + (size_t)NPTS * CH;           // N*64
  float*  hwb    = adstb + (size_t)NPTS * CH;           // N*64
  float*  outb   = hwb   + (size_t)NPTS * CH;           // N*64
  float*  ubuf   = outb  + (size_t)NPTS * CH;           // E*3
  float*  ybuf   = ubuf  + (size_t)NEDGE * 3;           // E*8

  // stats layout (doubles)
  double* S_BN1  = stats + 0;    double* Q_BN1 = stats + 64;
  double* S_POS  = stats + 128;  double* Q_POS = stats + 136;
  double* S_A1   = stats + 144;  double* Q_A1  = stats + 208;
  double* S_A2   = stats + 272;  double* Q_A2  = stats + 280;
  double* S_BN2  = stats + 288;  double* Q_BN2 = stats + 352;
  double* S_BN3  = stats + 416;  double* Q_BN3 = stats + 480;
  // param layout (floats): scale / shift pairs
  float* P_BN1S = params + 0,   *P_BN1H = params + 64;
  float* P_POSS = params + 128, *P_POSH = params + 136;
  float* P_A1S  = params + 160, *P_A1H  = params + 224;
  float* P_A2S  = params + 288, *P_A2H  = params + 296;
  float* P_BN2S = params + 304, *P_BN2H = params + 368;
  float* P_BN3S = params + 432, *P_BN3H = params + 496;

  const double invN = 1.0 / (double)NPTS;
  const double invE = 1.0 / (double)NEDGE;

  zero_stats<<<4, 256, 0, stream>>>(stats);

  // t = x @ W_in^T  (+ bn1 stats)
  gemm64<false><<<NPTS / 128, 256, 0, stream>>>(x, W_in, tbuf, nullptr, nullptr, S_BN1, Q_BN1);
  bn_finalize<<<1, 64, 0, stream>>>(S_BN1, Q_BN1, bn1_g, bn1_b, P_BN1S, P_BN1H, 64, invN);

  // h = relu(bn1(t)); a_src/a_dst/hW projections (WMMA)
  gemm64_triple<<<NPTS / 128, 256, 0, stream>>>(tbuf, P_BN1S, P_BN1H,
                                                src_w, src_b, dst_w, dst_b,
                                                lin_w, lin_b, asrcb, adstb, hwb);

  // pos MLP first layer + pos-BN stats
  pos_stats<<<NEDGE / 256, 256, 0, stream>>>(src, pos, pos_w1, pos_b1, ubuf, S_POS, Q_POS);
  bn_finalize<<<1, 64, 0, stream>>>(S_POS, Q_POS, pos_bn_g, pos_bn_b, P_POSS, P_POSH, 3, invE);

  // attn BN1 stats over a = a_src[src]-a_dst[dst]+delta
  attn1_stats<<<NEDGE / 256, 256, 0, stream>>>(src, asrcb, adstb, ubuf, P_POSS, P_POSH,
                                               pos_w2, pos_b2, S_A1, Q_A1);
  bn_finalize<<<1, 64, 0, stream>>>(S_A1, Q_A1, a1_g, a1_b, P_A1S, P_A1H, 64, invE);

  // y = relu(bn1(a)) @ attn_w1^T + b1  (+ attn BN2 stats)
  attn_mlp1<<<NEDGE / 256, 256, 0, stream>>>(src, asrcb, adstb, ubuf, P_POSS, P_POSH,
                                             pos_w2, pos_b2, P_A1S, P_A1H,
                                             attn_w1, attn_b1, ybuf, S_A2, Q_A2);
  bn_finalize<<<1, 64, 0, stream>>>(S_A2, Q_A2, a2_g, a2_b, P_A2S, P_A2H, 8, invE);

  // z -> softmax over 16 -> weighted aggregation (+ bn2 stats)
  softmax_agg<<<NEDGE / 256, 256, 0, stream>>>(src, ybuf, ubuf, P_POSS, P_POSH,
                                               pos_w2, pos_b2, P_A2S, P_A2H,
                                               attn_w2, attn_b2, hwb, outb, S_BN2, Q_BN2);
  bn_finalize<<<1, 64, 0, stream>>>(S_BN2, Q_BN2, bn2_g, bn2_b, P_BN2S, P_BN2H, 64, invN);

  // t3 = relu(bn2(out)) @ W_out^T  (+ bn3 stats)  -- reuses tbuf
  gemm64<true><<<NPTS / 128, 256, 0, stream>>>(outb, W_out, tbuf, P_BN2S, P_BN2H, S_BN3, Q_BN3);
  bn_finalize<<<1, 64, 0, stream>>>(S_BN3, Q_BN3, bn3_g, bn3_b, P_BN3S, P_BN3H, 64, invN);

  // out = relu(bn3(t3) + x)
  final_out<<<(NPTS * CH) / 256, 256, 0, stream>>>(tbuf, x, P_BN3S, P_BN3H, (float*)d_out);
}
